// TransformerBlock_81423989997621
// MI455X (gfx1250) — matrix operand order repaired and checked by reading, not compiled, not run on hardware
//
#include <hip/hip_runtime.h>
#include <hip/hip_bf16.h>
#include <math.h>

// ---------------------------------------------------------------------------
// Problem constants (from reference): B=2, S=2048, H=1024, NH=16, HD=64, I=4096
// ---------------------------------------------------------------------------
#define BB  2
#define SS  2048
#define HH  1024
#define NHH 16
#define HDD 64
#define II  4096
#define MS  (BB * SS)      // 4096 total token rows

typedef __bf16 bf16;
typedef __attribute__((ext_vector_type(8)))  bf16  v8bf;
typedef __attribute__((ext_vector_type(16))) bf16  v16bf;
typedef __attribute__((ext_vector_type(8)))  float v8f;
typedef __attribute__((ext_vector_type(4)))  unsigned int u32x4;
typedef __attribute__((ext_vector_type(8)))  int i32x8;
typedef __attribute__((ext_vector_type(4)))  int i32x4;

#define WMMA_BF16(a, b, c) \
  __builtin_amdgcn_wmma_f32_16x16x32_bf16(false, (a), false, (b), (short)0, (c), false, false)

// ---------------------------------------------------------------------------
// Tensor Data Mover: 2-D tile load global -> LDS, bf16 elements.
// Descriptor bitfields per CDNA5 ISA 8.3 (group0) / 8.4 (group1):
//   g0: [1:0]=count=1, [63:32]=lds_addr, [120:64]=global_addr, [127:126]=type=2
//   g1: [17:16]=data_size(1=2B), [79:48]=tensor_dim0, [111:80]=tensor_dim1,
//       [127:112]=tile_dim0, [143:128]=tile_dim1, [207:160]=tensor_dim0_stride
// Tile dims == tensor dims (our tiles never clip), stride0 in elements.
// This toolchain's builtin takes 6 args: groups 0..3 + extra i32x8 + cpol.
// Caller guards so exactly one wave per block issues the DMA (EXEC ignored).
// ---------------------------------------------------------------------------
__device__ __forceinline__ void tdm_load_2d(const void* gptr, void* lptr,
                                            unsigned tile0, unsigned tile1,
                                            unsigned long long stride0) {
  const unsigned long long ga = (unsigned long long)gptr;
  u32x4 g0;
  g0[0] = 1u;                                              // count=1, user D#
  g0[1] = (unsigned)(unsigned long long)lptr;              // LDS byte offset
  g0[2] = (unsigned)ga;
  g0[3] = ((unsigned)(ga >> 32) & 0x01FFFFFFu) | (2u << 30);
  i32x8 g1;
  g1[0] = 1 << 16;                                         // data_size = 2 bytes
  g1[1] = (int)((tile0 & 0xFFFFu) << 16);                  // tensor_dim0 lo
  g1[2] = (int)((tile0 >> 16) | ((tile1 & 0xFFFFu) << 16));// dim0 hi | dim1 lo
  g1[3] = (int)((tile1 >> 16) | (tile0 << 16));            // dim1 hi | tile_dim0
  g1[4] = (int)(tile1 & 0xFFFFu);                          // tile_dim1, tile_dim2=0
  g1[5] = (int)(stride0 & 0xFFFFFFFFull);                  // stride0 lo
  g1[6] = (int)((stride0 >> 32) & 0xFFFFull);              // stride0 hi
  g1[7] = 0;
  i32x4 z4 = {0, 0, 0, 0};
  i32x8 z8 = {0, 0, 0, 0, 0, 0, 0, 0};
  __builtin_amdgcn_tensor_load_to_lds(g0, g1, z4, z4, z8, 0);
}

// ---------------------------------------------------------------------------
// Fragment gathers from LDS, following CDNA5 ISA 7.12.2 (wave32):
//  A (16x32, 16-bit): lanes 0-15: M=lane, K=0..7 (v0-3) & 16..23 (v4-7);
//                     lanes 16-31: M=lane-16, K=+8.
//  B (32x16, 16-bit): lanes 0-15: N=lane, K=0..7 & 16..23; lanes 16-31: N=lane-16, K=+8.
// LDS is row-major [M][K] for A-style, [N][K] (transposed) for B-style, so
// each fragment is two contiguous ds_load_b128 per lane.
// ---------------------------------------------------------------------------
__device__ __forceinline__ v16bf fragA(const bf16* p, int row, int ldk, int k0) {
  const int lane = threadIdx.x & 31;
  const int r    = row + (lane & 15);
  const int koff = k0 + ((lane >> 4) << 3);
  const bf16* q  = p + r * ldk + koff;
  v8bf lo = *(const v8bf*)q;
  v8bf hi = *(const v8bf*)(q + 16);
  v16bf o;
#pragma unroll
  for (int i = 0; i < 8; ++i) { o[i] = lo[i]; o[i + 8] = hi[i]; }
  return o;
}

__device__ __forceinline__ v16bf fragB(const bf16* p, int row, int ldk, int k0) {
  const int lane = threadIdx.x & 31;
  const int r    = row + (lane & 15);
  const int koff = k0 + ((lane >> 4) << 3);
  const bf16* q  = p + r * ldk + koff;
  v8bf lo = *(const v8bf*)q;
  v8bf hi = *(const v8bf*)(q + 16);
  v16bf o;
#pragma unroll
  for (int i = 0; i < 8; ++i) { o[i] = lo[i]; o[i + 8] = hi[i]; }
  return o;
}

// ---------------------------------------------------------------------------
// Generic bf16 WMMA GEMM: C[M,N] = A[M,K] * B[K,N], f32 accumulate.
// Block: 256 threads (8 waves), tile 128x128; wave grid 4(M) x 2(N); each wave
// owns a 32x64 strip = 8 WMMA accumulators. Double-buffered LDS; the A tile is
// DMA'd by the TDM (wave 0 issues, all waves consume after s_wait_tensorcnt +
// barrier); the B tile is prefetched into registers before compute and
// transposed into LDS after compute.
// EPI: 0 = store f32; 1 = store f32 + residual R; 2 = exact GELU -> bf16.
// ---------------------------------------------------------------------------
template <int EPI>
__global__ __launch_bounds__(256)
void gemm_bf16(const bf16* __restrict__ A, const bf16* __restrict__ B,
               float* __restrict__ Cf, bf16* __restrict__ Cb,
               const float* __restrict__ R, int M, int N, int K) {
  __shared__ __attribute__((aligned(16))) bf16 lA[2][128 * 32];   // [m][k]
  __shared__ __attribute__((aligned(16))) bf16 lB[2][128 * 32];   // [n][k]
  const int tid  = threadIdx.x;
  const int lane = tid & 31;
  const int wave = tid >> 5;
  const int wm   = wave >> 1;      // 0..3
  const int wn   = wave & 1;       // 0..1
  const int bm   = blockIdx.y * 128;
  const int bn   = blockIdx.x * 128;

  v8f acc[2][4];
#pragma unroll
  for (int mt = 0; mt < 2; ++mt)
#pragma unroll
    for (int nt = 0; nt < 4; ++nt)
#pragma unroll
      for (int i = 0; i < 8; ++i) acc[mt][nt][i] = 0.0f;

  const int nK  = K >> 5;
  const int bk  = tid >> 3;            // B stage: k row 0..31
  const int bn0 = (tid & 7) << 4;      // B stage: n col 0..112 step 16

  // ---- prologue: stage tile 0 ----
  if (wave == 0)
    tdm_load_2d(&A[(size_t)bm * K], &lA[0][0], 32u, 128u, (unsigned long long)K);
  {
    const size_t gb = (size_t)bk * N + bn + bn0;
    v8bf t0 = *(const v8bf*)&B[gb];
    v8bf t1 = *(const v8bf*)&B[gb + 8];
#pragma unroll
    for (int i = 0; i < 8; ++i) {
      lB[0][(bn0 + i)     * 32 + bk] = t0[i];
      lB[0][(bn0 + 8 + i) * 32 + bk] = t1[i];
    }
  }
  if (wave == 0) __builtin_amdgcn_s_wait_tensorcnt(0);
  __syncthreads();

  for (int kt = 0; kt < nK; ++kt) {
    const int cur = kt & 1;
    const int nxt = cur ^ 1;
    const bool pf = (kt + 1) < nK;

    // prefetch next B tile into registers + issue next A tile DMA
    v8bf t0, t1;
    if (pf) {
      const size_t gb = (size_t)((kt + 1) * 32 + bk) * N + bn + bn0;
      t0 = *(const v8bf*)&B[gb];
      t1 = *(const v8bf*)&B[gb + 8];
      if (wave == 0)
        tdm_load_2d(&A[(size_t)bm * K + (kt + 1) * 32], &lA[nxt][0],
                    32u, 128u, (unsigned long long)K);
    }

    // compute on current buffers
    v16bf af[2], bfr[4];
#pragma unroll
    for (int mt = 0; mt < 2; ++mt) af[mt]  = fragA(lA[cur], wm * 32 + mt * 16, 32, 0);
#pragma unroll
    for (int nt = 0; nt < 4; ++nt) bfr[nt] = fragB(lB[cur], wn * 64 + nt * 16, 32, 0);
#pragma unroll
    for (int mt = 0; mt < 2; ++mt)
#pragma unroll
      for (int nt = 0; nt < 4; ++nt)
        acc[mt][nt] = WMMA_BF16(af[mt], bfr[nt], acc[mt][nt]);

    // commit prefetched B tile (transposed) into the next buffer
    if (pf) {
#pragma unroll
      for (int i = 0; i < 8; ++i) {
        lB[nxt][(bn0 + i)     * 32 + bk] = t0[i];
        lB[nxt][(bn0 + 8 + i) * 32 + bk] = t1[i];
      }
    }
    if (wave == 0) __builtin_amdgcn_s_wait_tensorcnt(0);
    __syncthreads();
  }

  // Epilogue. C/D layout: vgpr r -> M = r (+8 upper half-wave), N = lane&15.
  const int rb = (lane >> 4) << 3;
#pragma unroll
  for (int mt = 0; mt < 2; ++mt)
#pragma unroll
    for (int nt = 0; nt < 4; ++nt) {
      const int n     = bn + wn * 64 + nt * 16 + (lane & 15);
      const int mbase = bm + wm * 32 + mt * 16 + rb;
#pragma unroll
      for (int r = 0; r < 8; ++r) {
        const size_t idx = (size_t)(mbase + r) * N + n;
        float v = acc[mt][nt][r];
        if (EPI == 0) {
          Cf[idx] = v;
        } else if (EPI == 1) {
          Cf[idx] = v + R[idx];
        } else {
          float gl = 0.5f * v * (1.0f + erff(v * 0.70710678118654752f));
          Cb[idx] = (bf16)gl;
        }
      }
    }
}

// ---------------------------------------------------------------------------
// RMSNorm: one block (256 threads) per token row; fp32 reduce, bf16 output.
// ---------------------------------------------------------------------------
__global__ __launch_bounds__(256)
void rmsnorm_bf16(const float* __restrict__ x, const float* __restrict__ g,
                  bf16* __restrict__ o) {
  __shared__ float red[256];
  const int row   = blockIdx.x;
  const float* xr = x + (size_t)row * HH;
  float ss = 0.0f;
  for (int i = threadIdx.x; i < HH; i += 256) { float t = xr[i]; ss += t * t; }
  red[threadIdx.x] = ss;
  __syncthreads();
  for (int off = 128; off > 0; off >>= 1) {
    if (threadIdx.x < off) red[threadIdx.x] += red[threadIdx.x + off];
    __syncthreads();
  }
  const float norm = rsqrtf(red[0] / (float)HH + 1e-8f);
  for (int i = threadIdx.x; i < HH; i += 256)
    o[(size_t)row * HH + i] = (bf16)(g[i] * xr[i] * norm);
}

// ---------------------------------------------------------------------------
// Rotary embedding + split/scatter: qkv f32 [B,S,3,NH,HD] ->
// Q,K (rotated) and V, bf16 [B*NH, S, HD]. One thread per (b,h,s,d).
// ---------------------------------------------------------------------------
__global__ __launch_bounds__(256)
void rotary_split(const float* __restrict__ qkv, bf16* __restrict__ Qb,
                  bf16* __restrict__ Kb, bf16* __restrict__ Vb) {
  const size_t i = (size_t)blockIdx.x * 256 + threadIdx.x;
  if (i >= (size_t)BB * NHH * SS * HDD) return;
  const int d = (int)(i & 63);
  const int s = (int)((i >> 6) & (SS - 1));
  const int h = (int)((i >> 17) & (NHH - 1));
  const int b = (int)(i >> 21);

  const size_t hs  = (size_t)NHH * HDD;
  const size_t src = ((size_t)(b * SS + s) * 3 * NHH + h) * HDD + d;
  const float q  = qkv[src];
  const float k  = qkv[src + hs];
  const float v  = qkv[src + 2 * hs];
  const int   d2 = (d + 32) & 63;
  const float qp = qkv[src - d + d2];
  const float kp = qkv[src + hs - d + d2];
  const float sg = (d < 32) ? -1.0f : 1.0f;     // rotate_half sign

  const float inv = powf(10000.0f, -(float)(d & 31) / 32.0f);
  const float ang = (float)s * inv;
  const float c = cosf(ang), sn = sinf(ang);

  const size_t dst = ((size_t)(b * NHH + h) * SS + s) * HDD + d;
  Qb[dst] = (bf16)(q * c + sg * qp * sn);
  Kb[dst] = (bf16)(k * c + sg * kp * sn);
  Vb[dst] = (bf16)v;
}

// ---------------------------------------------------------------------------
// Flash attention. Grid: (S/64, B*NH). Block: 128 threads = 4 waves; each
// wave owns 16 query rows. Causal 32-key tiles. Q tile and K tiles are TDM
// loads (wave 0 issues); V is manually transposed into LDS. Scores and PV
// via bf16 WMMA; online softmax in fp32. Output bf16 in [B,S,H] layout.
// ---------------------------------------------------------------------------
__global__ __launch_bounds__(128)
void attn_flash(const bf16* __restrict__ Q, const bf16* __restrict__ Kc,
                const bf16* __restrict__ V, bf16* __restrict__ O) {
  __shared__ __attribute__((aligned(16))) bf16 lQ[64 * 64];        // [q][d]
  __shared__ __attribute__((aligned(16))) bf16 lK[32 * 64];        // [key][d]
  __shared__ __attribute__((aligned(16))) bf16 lV[64 * 32];        // [d][key]
  __shared__ __attribute__((aligned(16))) bf16 lP[4][16 * 32];     // per-wave P
  __shared__ float lS[4][16 * 32];                                  // per-wave scores
  __shared__ float lScale[4][16];
  __shared__ float lL[4][16];

  const int tid  = threadIdx.x;
  const int lane = tid & 31;
  const int w    = tid >> 5;
  const int bh   = blockIdx.y;
  const int b    = bh >> 4, h = bh & 15;
  const int q0   = blockIdx.x * 64;
  const bf16* Qh = Q  + (size_t)bh * SS * HDD;
  const bf16* Kh = Kc + (size_t)bh * SS * HDD;
  const bf16* Vh = V  + (size_t)bh * SS * HDD;

  // stage Q tile (64x64) via TDM; completion folded into first loop wait
  if (w == 0)
    tdm_load_2d(&Qh[(size_t)q0 * HDD], lQ, 64u, 64u, 64ull);

  v8f o[4];
#pragma unroll
  for (int t = 0; t < 4; ++t)
#pragma unroll
    for (int i = 0; i < 8; ++i) o[t][i] = 0.0f;
  float mrow = -1e30f, lrow = 0.0f;             // row stats live in lanes 0..15

  const int kbmax = (q0 + 63) >> 5;
  for (int kb = 0; kb <= kbmax; ++kb) {
    const int ks = kb * 32;
    __syncthreads();                            // everyone done with lK/lV
    if (w == 0)                                 // K tile [key][d] via TDM
      tdm_load_2d(&Kh[(size_t)ks * HDD], lK, 64u, 32u, 64ull);
    {   // V tile transposed into [d][key]
      const int r = tid >> 2;
      const int c = (tid & 3) * 16;
      const size_t gb = (size_t)(ks + r) * 64 + c;
      v8bf t0 = *(const v8bf*)&Vh[gb];
      v8bf t1 = *(const v8bf*)&Vh[gb + 8];
#pragma unroll
      for (int i = 0; i < 8; ++i) {
        lV[(c + i)     * 32 + r] = t0[i];
        lV[(c + 8 + i) * 32 + r] = t1[i];
      }
    }
    if (w == 0) __builtin_amdgcn_s_wait_tensorcnt(0);
    __syncthreads();

    // scores: 16(q) x 32(key), K-dim = HD = 64 -> 2 WMMA steps x 2 N-tiles
    v8f sc[2];
#pragma unroll
    for (int nt = 0; nt < 2; ++nt)
#pragma unroll
      for (int i = 0; i < 8; ++i) sc[nt][i] = 0.0f;
#pragma unroll
    for (int kc = 0; kc < 64; kc += 32) {
      v16bf aq = fragA(lQ, w * 16, 64, kc);
#pragma unroll
      for (int nt = 0; nt < 2; ++nt) {
        v16bf bk = fragB(lK, nt * 16, 64, kc);
        sc[nt] = WMMA_BF16(aq, bk, sc[nt]);
      }
    }

    {   // scale + causal mask -> per-wave LDS score tile
      const int rb = (lane >> 4) << 3;
#pragma unroll
      for (int nt = 0; nt < 2; ++nt) {
        const int kcol = nt * 16 + (lane & 15);
        const int key  = ks + kcol;
#pragma unroll
        for (int r = 0; r < 8; ++r) {
          const int qg = q0 + w * 16 + rb + r;
          float sv = sc[nt][r] * 0.125f;        // 1/sqrt(64)
          if (key > qg) sv = -1e30f;
          lS[w][(rb + r) * 32 + kcol] = sv;
        }
      }
    }

    // online softmax update: lane r < 16 owns query row r of this wave
    if (lane < 16) {
      float mnew = mrow;
#pragma unroll 4
      for (int j = 0; j < 32; ++j) mnew = fmaxf(mnew, lS[w][lane * 32 + j]);
      const float f = expf(mrow - mnew);
      float sum = 0.0f;
#pragma unroll 4
      for (int j = 0; j < 32; ++j) {
        const float p = expf(lS[w][lane * 32 + j] - mnew);
        sum += p;
        lP[w][lane * 32 + j] = (bf16)p;
      }
      lrow = lrow * f + sum;
      mrow = mnew;
      lScale[w][lane] = f;
    }

    {   // rescale O accumulators by per-row factor (wave-internal LDS b'cast)
      const int rb = (lane >> 4) << 3;
      float fr[8];
#pragma unroll
      for (int r = 0; r < 8; ++r) fr[r] = lScale[w][rb + r];
#pragma unroll
      for (int t = 0; t < 4; ++t)
#pragma unroll
        for (int r = 0; r < 8; ++r) o[t][r] *= fr[r];
    }

    // O += P @ V : A = P (16x32), B = V^T-in-LDS ([d][key]), 4 N-tiles of d
    v16bf ap = fragA(lP[w], 0, 32, 0);
#pragma unroll
    for (int t = 0; t < 4; ++t) {
      v16bf bv = fragB(lV, t * 16, 32, 0);
      o[t] = WMMA_BF16(ap, bv, o[t]);
    }
  }

  if (lane < 16) lL[w][lane] = lrow;
  const int rb = (lane >> 4) << 3;
  float li[8];
#pragma unroll
  for (int r = 0; r < 8; ++r) li[r] = 1.0f / lL[w][rb + r];
#pragma unroll
  for (int t = 0; t < 4; ++t) {
    const int d = t * 16 + (lane & 15);
#pragma unroll
    for (int r = 0; r < 8; ++r) {
      const int qg = q0 + w * 16 + rb + r;
      O[((size_t)(b * SS + qg)) * HH + h * HDD + d] = (bf16)(o[t][r] * li[r]);
    }
  }
}

// ---------------------------------------------------------------------------
__global__ __launch_bounds__(256)
void f32_to_bf16(const float* __restrict__ a, bf16* __restrict__ o, int n) {
  const int i = blockIdx.x * 256 + threadIdx.x;
  if (i < n) o[i] = (bf16)a[i];
}

// ---------------------------------------------------------------------------
extern "C" void kernel_launch(void* const* d_in, const int* in_sizes, int n_in,
                              void* d_out, int out_size, void* d_ws, size_t ws_size,
                              hipStream_t stream) {
  const float* x    = (const float*)d_in[0];
  // d_in[1] = attn_mask (causal; applied analytically)
  const float* g1   = (const float*)d_in[2];
  const float* g2   = (const float*)d_in[3];
  const float* Wqkv = (const float*)d_in[4];
  const float* Wout = (const float*)d_in[5];
  const float* W1   = (const float*)d_in[6];
  const float* W2   = (const float*)d_in[7];
  float* out = (float*)d_out;

  char*  ws  = (char*)d_ws;
  size_t off = 0;
  auto alloc = [&](size_t bytes) -> void* {
    void* p = ws + off;
    off = (off + bytes + 255) & ~(size_t)255;
    return p;
  };

  bf16*  hbf    = (bf16*)alloc((size_t)MS * HH * 2);
  float* qkvf   = (float*)alloc((size_t)MS * 3 * HH * 4);
  bf16*  qb     = (bf16*)alloc((size_t)MS * HH * 2);
  bf16*  kbuf   = (bf16*)alloc((size_t)MS * HH * 2);
  bf16*  vb     = (bf16*)alloc((size_t)MS * HH * 2);
  bf16*  attnb  = (bf16*)alloc((size_t)MS * HH * 2);
  float* x1     = (float*)alloc((size_t)MS * HH * 4);
  bf16*  h2b    = (bf16*)alloc((size_t)MS * HH * 2);
  bf16*  ub     = (bf16*)alloc((size_t)MS * II * 2);
  bf16*  wqkvb  = (bf16*)alloc((size_t)HH * 3 * HH * 2);
  bf16*  woutb  = (bf16*)alloc((size_t)HH * HH * 2);
  bf16*  w1b    = (bf16*)alloc((size_t)HH * II * 2);
  bf16*  w2b    = (bf16*)alloc((size_t)II * HH * 2);

  // weights -> bf16
  f32_to_bf16<<<(HH * 3 * HH + 255) / 256, 256, 0, stream>>>(Wqkv, wqkvb, HH * 3 * HH);
  f32_to_bf16<<<(HH * HH + 255) / 256, 256, 0, stream>>>(Wout, woutb, HH * HH);
  f32_to_bf16<<<(HH * II + 255) / 256, 256, 0, stream>>>(W1, w1b, HH * II);
  f32_to_bf16<<<(II * HH + 255) / 256, 256, 0, stream>>>(W2, w2b, II * HH);

  // h = rmsnorm(x, g1)
  rmsnorm_bf16<<<MS, 256, 0, stream>>>(x, g1, hbf);

  // qkv = h @ Wqkv  (f32 accumulate)
  gemm_bf16<0><<<dim3(3 * HH / 128, MS / 128), 256, 0, stream>>>(
      hbf, wqkvb, qkvf, nullptr, nullptr, MS, 3 * HH, HH);

  // rotary + split into per-head bf16 Q/K/V
  rotary_split<<<(BB * NHH * SS * HDD) / 256, 256, 0, stream>>>(qkvf, qb, kbuf, vb);

  // causal flash attention -> attnb [B,S,H] bf16
  attn_flash<<<dim3(SS / 64, BB * NHH), 128, 0, stream>>>(qb, kbuf, vb, attnb);

  // x1 = x + attn @ Wout
  gemm_bf16<1><<<dim3(HH / 128, MS / 128), 256, 0, stream>>>(
      attnb, woutb, x1, nullptr, x, MS, HH, HH);

  // h2 = rmsnorm(x1, g2)
  rmsnorm_bf16<<<MS, 256, 0, stream>>>(x1, g2, h2b);

  // u = gelu(h2 @ W1) -> bf16
  gemm_bf16<2><<<dim3(II / 128, MS / 128), 256, 0, stream>>>(
      h2b, w1b, nullptr, ub, nullptr, MS, II, HH);

  // out = x1 + u @ W2
  gemm_bf16<1><<<dim3(HH / 128, MS / 128), 256, 0, stream>>>(
      ub, w2b, out, nullptr, x1, MS, HH, II);
}
